// MLA_9620726743656
// MI455X (gfx1250) — matrix operand order repaired and checked by reading, not compiled, not run on hardware
//
#include <hip/hip_runtime.h>

// ---------------------------------------------------------------------------
// MLA forward for MI455X (gfx1250, wave32, WMMA).
//
// Algebraic optimization: the reference has no softmax, so
//     out = (K Q^T) V / sqrt(D) == K (Q^T V) / sqrt(D)
// turning the S x S attention (68 GFLOP + 1 GiB scores tensor) into two
// 128-wide GEMMs (~4 GFLOP). The remaining ~130 GFLOP of projections run as
// bf16 WMMA 16x16x32 with f32 accumulation.
//
// v2 changes (driven by round-1 asm):
//  * All B operands pre-transposed in global memory (weights via a tiled
//    transpose-convert kernel; T^T written directly by the QtV epilogue),
//    so both A and B LDS staging are pure b128 vector copies -- the 16x
//    scalar ds_store_b16 scatter is gone.
//  * Explicit double-buffered LDS pipeline in the GEMM: global loads for
//    tile i+1 issue into registers while tile i computes; one barrier per
//    K-step instead of two; HBM latency overlaps WMMA.
//  * Prefetch locality raised so global_prefetch_b8 targets near caches.
// ---------------------------------------------------------------------------

#define NHEADS 16
#define DHEAD  128
#define DROPE  64

typedef __bf16 bf16_t;
typedef bf16_t bf16x16 __attribute__((ext_vector_type(16)));
typedef float  f32x8   __attribute__((ext_vector_type(8)));

union FragAB {
    bf16x16        v;
    uint4          q[2];
    unsigned short h[16];
};

__device__ __forceinline__ unsigned short f32_to_bf16_rne(float f) {
    unsigned int u = __float_as_uint(f);
    u += 0x7FFFu + ((u >> 16) & 1u);   // round-to-nearest-even
    return (unsigned short)(u >> 16);
}

__device__ __forceinline__ f32x8 zero8() {
    f32x8 z = {0.f, 0.f, 0.f, 0.f, 0.f, 0.f, 0.f, 0.f};
    return z;
}

__device__ __forceinline__ f32x8 wmma_bf16(const FragAB& a, const FragAB& b, f32x8 c) {
#if defined(__gfx1250__)
    // 8 args: (neg_a, A, neg_b, B, c_mod, C, reuse_a, reuse_b)
    return __builtin_amdgcn_wmma_f32_16x16x32_bf16(false, a.v, false, b.v,
                                                   (short)0, c, false, false);
#else
    return c;  // host pass never executes device code
#endif
}

// ---------------------------------------------------------------------------
// f32 -> bf16 conversion (grid-stride), for the A-side activation x
// ---------------------------------------------------------------------------
__global__ void cvt_f32_to_bf16(const float* __restrict__ src,
                                unsigned short* __restrict__ dst,
                                long long n) {
    long long i = (long long)blockIdx.x * blockDim.x + threadIdx.x;
    long long stride = (long long)gridDim.x * blockDim.x;
    for (; i < n; i += stride) dst[i] = f32_to_bf16_rne(src[i]);
}

// ---------------------------------------------------------------------------
// Tiled transpose + f32->bf16: src (K,N) f32 row-major -> dst (N,K) bf16
// row-major. 32x32 tile via LDS (33-stride kills bank conflicts).
// Run once per weight; makes GEMM B-staging a straight b128 copy.
// ---------------------------------------------------------------------------
__global__ __launch_bounds__(256)
void cvt_transpose_bf16(const float* __restrict__ src,
                        unsigned short* __restrict__ dst,
                        int K, int N) {
    __shared__ unsigned short tile[32][33];
    const int k0 = blockIdx.y * 32;
    const int n0 = blockIdx.x * 32;
    const int tx = threadIdx.x & 31;
    const int ty = threadIdx.x >> 5;  // 0..7
#pragma unroll
    for (int i = 0; i < 4; ++i) {
        const int k = ty + i * 8;
        tile[k][tx] = f32_to_bf16_rne(src[(size_t)(k0 + k) * N + n0 + tx]);
    }
    __syncthreads();
#pragma unroll
    for (int i = 0; i < 4; ++i) {
        const int n = ty + i * 8;
        dst[(size_t)(n0 + n) * K + k0 + tx] = tile[tx][n];
    }
}

// ---------------------------------------------------------------------------
// Generic bf16 GEMM, C = A(MxK) * B(KxN), f32 accumulate.
// A row-major (lda), B PRE-TRANSPOSED: Bt is (N,K) row-major (ldb = its row
// stride). 128x128 block tile, BK=32, 256 threads = 8 waves, wave tile 64x32
// (4x2 WMMA tiles). Double-buffered LDS pipeline: one barrier per K-step.
// Batched via grid.z (z = zo*inner + zi) for the per-(b,h) k@T GEMM.
// ---------------------------------------------------------------------------
__global__ __launch_bounds__(256)
void gemm_bf16_128x128(const unsigned short* __restrict__ Ab,
                       const unsigned short* __restrict__ Btb,
                       float* Cf, unsigned short* Cb,
                       int M, int N, int K,
                       int lda, int ldb, int ldc,
                       int batch_inner,
                       long long sAo, long long sAi,
                       long long sBo, long long sBi,
                       long long sCo, long long sCi) {
    __shared__ __align__(16) unsigned short As[2][128 * 32];  // [m][k]
    __shared__ __align__(16) unsigned short Bs[2][128 * 32];  // [n][k]

    const int z  = blockIdx.z;
    const int zo = z / batch_inner;
    const int zi = z % batch_inner;
    const unsigned short* A  = Ab  + zo * sAo + zi * sAi;
    const unsigned short* Bt = Btb + zo * sBo + zi * sBi;
    const long long coff = zo * sCo + zi * sCi;

    const int m0 = blockIdx.y * 128;
    const int n0 = blockIdx.x * 128;
    const int t    = threadIdx.x;
    const int lane = t & 31;
    const int wid  = t >> 5;
    const int waveM = (wid >> 2) * 64;  // 2 wave rows
    const int waveN = (wid & 3) * 32;   // 4 wave cols

    f32x8 acc[4][2];
#pragma unroll
    for (int i = 0; i < 4; ++i)
#pragma unroll
        for (int j = 0; j < 2; ++j) acc[i][j] = zero8();

    // staging map: each thread copies 32B of A and 32B of B per K-step
    const int row = t >> 1, seg = t & 1;
    const unsigned short* Abase  = A  + (size_t)(m0 + row) * lda + seg * 16;
    const unsigned short* Btbase = Bt + (size_t)(n0 + row) * ldb + seg * 16;

    uint4 ra0, ra1, rb0, rb1;
    auto gload = [&](int k0) {
        const uint4* ap = reinterpret_cast<const uint4*>(Abase + k0);
        ra0 = ap[0];
        ra1 = ap[1];
        const uint4* bp = reinterpret_cast<const uint4*>(Btbase + k0);
        rb0 = bp[0];
        rb1 = bp[1];
    };
    auto lstore = [&](int buf) {
        uint4* ad = reinterpret_cast<uint4*>(&As[buf][row * 32 + seg * 16]);
        ad[0] = ra0;
        ad[1] = ra1;
        uint4* bd = reinterpret_cast<uint4*>(&Bs[buf][row * 32 + seg * 16]);
        bd[0] = rb0;
        bd[1] = rb1;
    };
    auto compute = [&](int buf) {
        // fragment builds per documented 16-bit A/B VGPR layouts
        const int klo = (lane >> 4) * 8;    // A: lanes 16-31 hold K=8..15 / 24..31
        const int kbb = (lane >> 4) * 8;    // B: lanes 16-31 hold K=8..15 / 24..31
        FragAB fa[4], fb[2];
#pragma unroll
        for (int i = 0; i < 4; ++i) {
            const int m = waveM + i * 16 + (lane & 15);
            fa[i].q[0] = *reinterpret_cast<const uint4*>(&As[buf][m * 32 + klo]);
            fa[i].q[1] = *reinterpret_cast<const uint4*>(&As[buf][m * 32 + 16 + klo]);
        }
#pragma unroll
        for (int j = 0; j < 2; ++j) {
            const int n = waveN + j * 16 + (lane & 15);
            fb[j].q[0] = *reinterpret_cast<const uint4*>(&Bs[buf][n * 32 + kbb]);
            fb[j].q[1] = *reinterpret_cast<const uint4*>(&Bs[buf][n * 32 + 16 + kbb]);
        }
#pragma unroll
        for (int i = 0; i < 4; ++i)
#pragma unroll
            for (int j = 0; j < 2; ++j) acc[i][j] = wmma_bf16(fa[i], fb[j], acc[i][j]);
    };

    // ---- double-buffered pipeline: one barrier per K-step ----
    gload(0);
    lstore(0);
    __syncthreads();
    int buf = 0;
    for (int k0 = 32; k0 < K; k0 += 32) {
        gload(k0);                       // next tile -> registers (in flight)
        if (k0 + 32 < K) {               // look one tile further ahead
            __builtin_prefetch(Abase + k0 + 32, 0, 3);
            __builtin_prefetch(Btbase + k0 + 32, 0, 3);
        }
        compute(buf);                    // WMMA on current tile
        lstore(buf ^ 1);                 // staged registers -> other buffer
        __syncthreads();
        buf ^= 1;
    }
    compute(buf);

    // ---- epilogue: C/D layout: VGPR v -> M = v (+8 for lanes 16-31), N = lane&15
    const int mo = (lane >> 4) << 3;
    const int nn = lane & 15;
#pragma unroll
    for (int i = 0; i < 4; ++i)
#pragma unroll
        for (int j = 0; j < 2; ++j)
#pragma unroll
            for (int v = 0; v < 8; ++v) {
                const int gm = m0 + waveM + i * 16 + mo + v;
                const int gn = n0 + waveN + j * 16 + nn;
                const float val = acc[i][j][v];
                const size_t off = (size_t)coff + (size_t)gm * ldc + gn;
                if (Cf) Cf[off] = val;
                if (Cb) Cb[off] = f32_to_bf16_rne(val);
            }
}

// ---------------------------------------------------------------------------
// T^T[b,h] = transpose( (1/sqrt(128)) * Q_h^T V_h )  (128x128, bf16 out)
// One workgroup per (b,h); stages 32 s-rows of q and v into LDS transposed
// ([d][s]) so A = Q^T fragments are contiguous 16B reads. Epilogue stores
// T transposed so the downstream k@T GEMM consumes it as a pre-transposed B.
// ---------------------------------------------------------------------------
__global__ __launch_bounds__(256)
void qtv_bf16(const unsigned short* __restrict__ qb,   // (B,S,NH,DH) bf16
              const unsigned short* __restrict__ vb,   // (B,S,NH,DH) bf16
              unsigned short* __restrict__ Tt,         // (B*NH,128,128) bf16, transposed
              int S) {
    __shared__ __align__(16) unsigned short Qs[128 * 32];  // [d1][s]
    __shared__ __align__(16) unsigned short Vs[128 * 32];  // [d2][s]

    const int z = blockIdx.x;
    const int b = z >> 4, h = z & 15;
    const unsigned short* qp = qb + (size_t)b * S * 2048 + h * 128;
    const unsigned short* vp = vb + (size_t)b * S * 2048 + h * 128;

    const int t    = threadIdx.x;
    const int lane = t & 31;
    const int wid  = t >> 5;
    const int waveM = (wid >> 2) * 64;
    const int waveN = (wid & 3) * 32;

    f32x8 acc[4][2];
#pragma unroll
    for (int i = 0; i < 4; ++i)
#pragma unroll
        for (int j = 0; j < 2; ++j) acc[i][j] = zero8();

    const int ls = t >> 3, seg = t & 7;  // 32 s-rows x 8 x 16-elem segments

    for (int s0 = 0; s0 < S; s0 += 32) {
        FragAB tq, tv;
        const uint4* qs = reinterpret_cast<const uint4*>(qp + (size_t)(s0 + ls) * 2048 + seg * 16);
        const uint4* vs = reinterpret_cast<const uint4*>(vp + (size_t)(s0 + ls) * 2048 + seg * 16);
        tq.q[0] = qs[0]; tq.q[1] = qs[1];
        tv.q[0] = vs[0]; tv.q[1] = vs[1];
#pragma unroll
        for (int e = 0; e < 16; ++e) {
            Qs[(seg * 16 + e) * 32 + ls] = tq.h[e];
            Vs[(seg * 16 + e) * 32 + ls] = tv.h[e];
        }
        __syncthreads();

        const int klo = (lane >> 4) * 8;
        const int kbb = (lane >> 4) * 8;
        FragAB fa[4], fb[2];
#pragma unroll
        for (int i = 0; i < 4; ++i) {
            const int m = waveM + i * 16 + (lane & 15);
            fa[i].q[0] = *reinterpret_cast<const uint4*>(&Qs[m * 32 + klo]);
            fa[i].q[1] = *reinterpret_cast<const uint4*>(&Qs[m * 32 + 16 + klo]);
        }
#pragma unroll
        for (int j = 0; j < 2; ++j) {
            const int n = waveN + j * 16 + (lane & 15);
            fb[j].q[0] = *reinterpret_cast<const uint4*>(&Vs[n * 32 + kbb]);
            fb[j].q[1] = *reinterpret_cast<const uint4*>(&Vs[n * 32 + 16 + kbb]);
        }
#pragma unroll
        for (int i = 0; i < 4; ++i)
#pragma unroll
            for (int j = 0; j < 2; ++j) acc[i][j] = wmma_bf16(fa[i], fb[j], acc[i][j]);
        __syncthreads();
    }

    const float scale = 0.08838834764831845f;  // 1/sqrt(128)
    const int mo = (lane >> 4) << 3;
    const int nn = lane & 15;
#pragma unroll
    for (int i = 0; i < 4; ++i)
#pragma unroll
        for (int j = 0; j < 2; ++j)
#pragma unroll
            for (int v = 0; v < 8; ++v) {
                const int gm = waveM + i * 16 + mo + v;   // d1 (k of next GEMM)
                const int gn = waveN + j * 16 + nn;       // d2 (n of next GEMM)
                // store transposed: Tt[n][k]
                Tt[(size_t)z * 16384 + (size_t)gn * 128 + gm] =
                    f32_to_bf16_rne(acc[i][j][v] * scale);
            }
}

// ---------------------------------------------------------------------------
// Fused RoPE + head assembly.
//   - k_rope f32 output (reference output #3)
//   - assembled k, q bf16 (B,S,NH,128): [k_c(64) | k_rope(64)]
// One thread per (b,s,h,d), d in [0,64).
// ---------------------------------------------------------------------------
__global__ void rope_assemble(const float* __restrict__ krf,          // (B,S,NH*64) f32
                              const float* __restrict__ qrf,          // (B,S,NH*64) f32
                              const unsigned short* __restrict__ kcb, // (B,S,NH*64) bf16
                              const unsigned short* __restrict__ qcb, // (B,S,NH*64) bf16
                              float* __restrict__ krope_out,          // (B,S,NH,64) f32
                              unsigned short* __restrict__ kb,        // (B,S,NH,128) bf16
                              unsigned short* __restrict__ qb,        // (B,S,NH,128) bf16
                              int S) {
    const long long tid = (long long)blockIdx.x * blockDim.x + threadIdx.x;
    const long long total = (long long)2 * S * NHEADS * DROPE;
    if (tid >= total) return;

    const int d = (int)(tid & 63);
    const long long r  = tid >> 6;       // (b*S+s)*16 + h
    const int h        = (int)(r & 15);
    const long long bs = r >> 4;         // b*S + s
    const int s        = (int)(bs % S);

    const size_t rbase = (size_t)bs * (NHEADS * DROPE) + (size_t)h * DROPE;
    const size_t obase = (size_t)bs * 2048 + (size_t)h * 128;

    float kv, qv;
    if (d < 32) {
        // inv_freq[j] = 10000^(-j/16), rotation = concat([freqs, freqs])
        const float inv = __powf(10000.0f, -(float)(d & 15) / 16.0f);
        const float ang = ((float)s / 40.0f) * inv;  // ROPE_SCALE = 40
        float sn, cs;
        __sincosf(ang, &sn, &cs);
        const float k0 = krf[rbase + d];
        const float kp = (d < 16) ? -krf[rbase + d + 16] : krf[rbase + d - 16];
        kv = k0 * cs + kp * sn;
        const float q0 = qrf[rbase + d];
        const float qp = (d < 16) ? -qrf[rbase + d + 16] : qrf[rbase + d - 16];
        qv = q0 * cs + qp * sn;
    } else {
        kv = krf[rbase + d];  // pass-through base half
        qv = qrf[rbase + d];
    }

    krope_out[rbase + d] = kv;
    kb[obase + 64 + d] = f32_to_bf16_rne(kv);
    qb[obase + 64 + d] = f32_to_bf16_rne(qv);
    kb[obase + d] = kcb[rbase + d];  // k_c occupies dims [0,64)
    qb[obase + d] = qcb[rbase + d];
}

// ---------------------------------------------------------------------------
// Host launcher
// ---------------------------------------------------------------------------
extern "C" void kernel_launch(void* const* d_in, const int* in_sizes, int n_in,
                              void* d_out, int out_size, void* d_ws, size_t ws_size,
                              hipStream_t stream) {
    (void)in_sizes; (void)n_in; (void)out_size; (void)ws_size;

    const int Bb = 2, S = 2048, H = 2048, CD = 1024, UP = 1024;
    const long long BS = (long long)Bb * S;  // 4096

    const float* x    = (const float*)d_in[0];
    const float* Wdkv = (const float*)d_in[1];
    const float* Wdq  = (const float*)d_in[2];
    const float* Wuk  = (const float*)d_in[3];
    const float* Wuv  = (const float*)d_in[4];
    const float* Wuq  = (const float*)d_in[5];
    const float* Wkr  = (const float*)d_in[6];
    const float* Wqr  = (const float*)d_in[7];
    const float* Wout = (const float*)d_in[8];

    // d_out: output (2,2048,2048) | c_kv (2,2048,1024) | k_rope (2,2048,16,64)
    float* out_f   = (float*)d_out;
    float* ckv_f   = out_f + (size_t)BS * H;
    float* krope_f = ckv_f + (size_t)BS * CD;

    // workspace carve-out
    char* w = (char*)d_ws;
    auto alloc = [&](size_t bytes) -> void* {
        void* p = (void*)w;
        w += (bytes + 255) & ~(size_t)255;
        return p;
    };
    auto* xb     = (unsigned short*)alloc((size_t)BS * H * 2);
    auto* Wdkv_t = (unsigned short*)alloc((size_t)H * CD * 2);   // (N,K) bf16
    auto* Wdq_t  = (unsigned short*)alloc((size_t)H * CD * 2);
    auto* Wuk_t  = (unsigned short*)alloc((size_t)CD * UP * 2);
    auto* Wuv_t  = (unsigned short*)alloc((size_t)CD * (NHEADS * DHEAD) * 2);
    auto* Wuq_t  = (unsigned short*)alloc((size_t)CD * UP * 2);
    auto* Wkr_t  = (unsigned short*)alloc((size_t)H * (NHEADS * DROPE) * 2);
    auto* Wqr_t  = (unsigned short*)alloc((size_t)UP * (NHEADS * DROPE) * 2);
    auto* Wout_t = (unsigned short*)alloc((size_t)(NHEADS * DHEAD) * H * 2);
    auto* ckv_b  = (unsigned short*)alloc((size_t)BS * CD * 2);
    auto* cq_b   = (unsigned short*)alloc((size_t)BS * CD * 2);
    auto* qc_b   = (unsigned short*)alloc((size_t)BS * UP * 2);
    auto* kc_b   = (unsigned short*)alloc((size_t)BS * UP * 2);
    auto* v_b    = (unsigned short*)alloc((size_t)BS * (NHEADS * DHEAD) * 2);
    auto* krf    = (float*)alloc((size_t)BS * (NHEADS * DROPE) * 4);
    auto* qrf    = (float*)alloc((size_t)BS * (NHEADS * DROPE) * 4);
    auto* k_b    = (unsigned short*)alloc((size_t)BS * (NHEADS * DHEAD) * 2);
    auto* q_b    = (unsigned short*)alloc((size_t)BS * (NHEADS * DHEAD) * 2);
    auto* Tt_b   = (unsigned short*)alloc((size_t)Bb * NHEADS * DHEAD * DHEAD * 2);
    auto* out_b  = (unsigned short*)alloc((size_t)BS * (NHEADS * DHEAD) * 2);

    // x: plain convert (A operand, row-major)
    cvt_f32_to_bf16<<<4096, 256, 0, stream>>>(x, xb, BS * (long long)H);

    // weights: transpose-convert to (N,K) bf16 for vectorized B staging
    auto cvtT = [&](const float* s, unsigned short* d, int K, int N) {
        dim3 grid(N / 32, K / 32);
        cvt_transpose_bf16<<<grid, 256, 0, stream>>>(s, d, K, N);
    };
    cvtT(Wdkv, Wdkv_t, H, CD);
    cvtT(Wdq,  Wdq_t,  H, CD);
    cvtT(Wuk,  Wuk_t,  CD, UP);
    cvtT(Wuv,  Wuv_t,  CD, NHEADS * DHEAD);
    cvtT(Wuq,  Wuq_t,  CD, UP);
    cvtT(Wkr,  Wkr_t,  H, NHEADS * DROPE);
    cvtT(Wqr,  Wqr_t,  UP, NHEADS * DROPE);
    cvtT(Wout, Wout_t, NHEADS * DHEAD, H);

    auto gemm = [&](const unsigned short* A, const unsigned short* Bt, float* Cf,
                    unsigned short* Cb, int M, int N, int K, int lda, int ldb, int ldc,
                    int batch, int inner,
                    long long sAo, long long sAi, long long sBo, long long sBi,
                    long long sCo, long long sCi) {
        dim3 grid(N / 128, M / 128, batch);
        gemm_bf16_128x128<<<grid, 256, 0, stream>>>(A, Bt, Cf, Cb, M, N, K, lda, ldb, ldc,
                                                    inner, sAo, sAi, sBo, sBi, sCo, sCi);
    };

    // projections (ldb = K: transposed-B row stride)
    gemm(xb,    Wdkv_t, ckv_f, ckv_b, (int)BS, CD, H, H, H, CD, 1, 1, 0, 0, 0, 0, 0, 0);
    gemm(xb,    Wdq_t,  nullptr, cq_b, (int)BS, CD, H, H, H, CD, 1, 1, 0, 0, 0, 0, 0, 0);
    gemm(xb,    Wkr_t,  krf, nullptr, (int)BS, NHEADS * DROPE, H, H, H, NHEADS * DROPE,
         1, 1, 0, 0, 0, 0, 0, 0);
    gemm(cq_b,  Wuq_t,  nullptr, qc_b, (int)BS, UP, CD, CD, CD, UP, 1, 1, 0, 0, 0, 0, 0, 0);
    gemm(ckv_b, Wuk_t,  nullptr, kc_b, (int)BS, UP, CD, CD, CD, UP, 1, 1, 0, 0, 0, 0, 0, 0);
    gemm(ckv_b, Wuv_t,  nullptr, v_b, (int)BS, NHEADS * DHEAD, CD, CD, CD, NHEADS * DHEAD,
         1, 1, 0, 0, 0, 0, 0, 0);
    gemm(qc_b,  Wqr_t,  qrf, nullptr, (int)BS, NHEADS * DROPE, UP, UP, UP, NHEADS * DROPE,
         1, 1, 0, 0, 0, 0, 0, 0);

    // RoPE + head assembly (writes k_rope output + bf16 k, q)
    {
        const long long total = (long long)Bb * S * NHEADS * DROPE;  // 4,194,304
        const int blocks = (int)((total + 255) / 256);
        rope_assemble<<<blocks, 256, 0, stream>>>(krf, qrf, kc_b, qc_b, krope_f, k_b, q_b, S);
    }

    // T^T[b,h] = transpose(Q_h^T V_h / sqrt(128))
    qtv_bf16<<<Bb * NHEADS, 256, 0, stream>>>(q_b, v_b, Tt_b, S);

    // out[b,:,h,:] = k[b,:,h,:] @ T[b,h]  (batched over 32 (b,h) pairs; B = Tt)
    gemm(k_b, Tt_b, nullptr, out_b, S, DHEAD, DHEAD, NHEADS * DHEAD, DHEAD, NHEADS * DHEAD,
         Bb * NHEADS, NHEADS,
         (long long)S * NHEADS * DHEAD, DHEAD,                          // A strides (b,h)
         (long long)NHEADS * DHEAD * DHEAD, (long long)DHEAD * DHEAD,   // Tt strides
         (long long)S * NHEADS * DHEAD, DHEAD);                         // C strides

    // output = out @ W_out -> f32 result
    gemm(out_b, Wout_t, out_f, nullptr, (int)BS, H, NHEADS * DHEAD, NHEADS * DHEAD,
         NHEADS * DHEAD, H, 1, 1, 0, 0, 0, 0, 0, 0);
}
